// MultiHeadAttention_39994735460719
// MI455X (gfx1250) — matrix operand order repaired and checked by reading, not compiled, not run on hardware
//
#include <hip/hip_runtime.h>
#include <hip/hip_bf16.h>

// ---------------------------------------------------------------------------
// CDNA5 (gfx1250) fused multi-head attention:
//   1) f32 -> bf16 conversion (vectorized)
//   2) Q/K/V projections: bf16 WMMA GEMM (Y = X*W^T + b) -> [B,H,S,Dh] bf16
//   3) flash attention (online softmax), v_wmma_f32_16x16x32_bf16 for QK^T
//      and P*V; K tiles staged via the Tensor Data Mover (tensor_load_to_lds
//      + s_wait_tensorcnt) when the builtin is available
//   4) output projection GEMM -> f32 d_out
// ---------------------------------------------------------------------------

#define MHA_S  4096
#define MHA_D  1024
#define MHA_H  16
#define MHA_DH 64

typedef __bf16 v16bf __attribute__((ext_vector_type(16)));
typedef float  v8f   __attribute__((ext_vector_type(8)));
typedef unsigned int u32x4 __attribute__((ext_vector_type(4)));
typedef int          i32x8 __attribute__((ext_vector_type(8)));
typedef int          i32x4 __attribute__((ext_vector_type(4)));

#if __has_builtin(__builtin_amdgcn_tensor_load_to_lds)
#define MHA_TDM 1
#else
#define MHA_TDM 0
#endif

union FragU {
    uint4 u[2];
    v16bf v;
};

__device__ __forceinline__ v8f wmma_bf16(v16bf a, v16bf b, v8f c) {
    // (neg_a, A, neg_b, B, c_mod, C, reuse_a, reuse_b)
    return __builtin_amdgcn_wmma_f32_16x16x32_bf16(false, a, false, b,
                                                   (short)0, c, false, false);
}

// 16-bit A-matrix 16x32 fragment (interleaved layout):
//   lanes 0-15 : row = rowBase+lane,    K = 0..7  and 16..23
//   lanes 16-31: row = rowBase+lane-16, K = 8..15 and 24..31
__device__ __forceinline__ v16bf ld_fragA(const __bf16* base, int strideB,
                                          int rowBase, int kByteOff, int lane) {
    const char* p = (const char*)base +
                    (size_t)(rowBase + (lane & 15)) * strideB + kByteOff +
                    ((lane & 16) ? 16 : 0);
    FragU f;
    f.u[0] = *(const uint4*)(p);
    f.u[1] = *(const uint4*)(p + 32);
    return f.v;
}

// 16-bit B-matrix 32x16 fragment: column (= N) stored contiguously along K:
//   lanes 0-15 : col = colBase+lane,    K = 0..7  and 16..23
//   lanes 16-31: col = colBase+lane-16, K = 8..15 and 24..31
__device__ __forceinline__ v16bf ld_fragB(const __bf16* base, int strideB,
                                          int colBase, int kByteOff, int lane) {
    const char* p = (const char*)base +
                    (size_t)(colBase + (lane & 15)) * strideB + kByteOff +
                    ((lane & 16) ? 16 : 0);
    FragU f;
    f.u[0] = *(const uint4*)(p);
    f.u[1] = *(const uint4*)(p + 32);
    return f.v;
}

// ---------------------------------------------------------------------------
// f32 -> bf16 conversion, 4 elements/thread/iter (b128 load, b64 store)
// ---------------------------------------------------------------------------
__global__ void mha_cvt_bf16_kernel(const float4* __restrict__ x,
                                    __bf16* __restrict__ y, size_t n4) {
    size_t i = (size_t)blockIdx.x * blockDim.x + threadIdx.x;
    size_t stride = (size_t)gridDim.x * blockDim.x;
    for (; i < n4; i += stride) {
        float4 f = x[i];
        union { __bf16 h[4]; uint2 u; } o;
        o.h[0] = (__bf16)f.x;
        o.h[1] = (__bf16)f.y;
        o.h[2] = (__bf16)f.z;
        o.h[3] = (__bf16)f.w;
        *(uint2*)(y + 4 * i) = o.u;
    }
}

// ---------------------------------------------------------------------------
// Y[M,N] = X[M,K] * W[N,K]^T + bias   (bf16 inputs, f32 accumulate)
// mode 0: write bf16 in [B, H, S, Dh] head layout   (N = H*Dh, M = B*S)
// mode 1: write f32 row-major [M, N]
// Tile: 128(M) x 64(N), K-step 32; 8 waves, each wave owns a 32x32 sub-tile
// ---------------------------------------------------------------------------
#define GM_BM 128
#define GM_BN 64
#define GM_BK 32

__global__ __launch_bounds__(256) void mha_gemm_xwt_kernel(
    const __bf16* __restrict__ X, const __bf16* __restrict__ W,
    const float* __restrict__ bias, void* __restrict__ Yout,
    int M, int N, int K, int mode) {
    __shared__ __bf16 As[GM_BM * GM_BK];  // row-major, row stride 64B
    __shared__ __bf16 Bs[GM_BN * GM_BK];  // per-column contiguous K, stride 64B

    const int tid  = threadIdx.x;
    const int lane = tid & 31;
    const int wave = tid >> 5;
    const int nBlocks = N / GM_BN;
    const int m0 = (blockIdx.x / nBlocks) * GM_BM;
    const int n0 = (blockIdx.x % nBlocks) * GM_BN;
    const int wm = (wave & 3) * 32;   // wave M offset in tile
    const int wn = (wave >> 2) * 32;  // wave N offset in tile

    v8f acc00 = {}, acc01 = {}, acc10 = {}, acc11 = {};

    const int ar = tid >> 1;            // 0..127
    const int ac = (tid & 1) * 16;      // 0 or 16
    const int bn = tid >> 2;            // 0..63
    const int bc = (tid & 3) * 8;       // 0,8,16,24

    for (int kk = 0; kk < K; kk += GM_BK) {
        // stage X panel 128x32
        {
            const uint4* src = (const uint4*)(X + (size_t)(m0 + ar) * K + kk + ac);
            uint4* dst = (uint4*)(As + ar * GM_BK + ac);
            dst[0] = src[0];
            dst[1] = src[1];
        }
        // stage W panel 64x32 (B column n = W row n, contiguous along K)
        {
            const uint4* src = (const uint4*)(W + (size_t)(n0 + bn) * K + kk + bc);
            uint4* dst = (uint4*)(Bs + bn * GM_BK + bc);
            dst[0] = src[0];
        }
        if (kk + GM_BK < K) {  // prefetch next panels -> global_prefetch_b8
            __builtin_prefetch(X + (size_t)(m0 + ar) * K + kk + GM_BK + ac, 0, 1);
            __builtin_prefetch(W + (size_t)(n0 + bn) * K + kk + GM_BK + bc, 0, 1);
        }
        __syncthreads();

        v16bf a0 = ld_fragA(As, 64, wm, 0, lane);
        v16bf a1 = ld_fragA(As, 64, wm + 16, 0, lane);
        v16bf b0 = ld_fragB(Bs, 64, wn, 0, lane);
        v16bf b1 = ld_fragB(Bs, 64, wn + 16, 0, lane);

        acc00 = wmma_bf16(a0, b0, acc00);
        acc01 = wmma_bf16(a0, b1, acc01);
        acc10 = wmma_bf16(a1, b0, acc10);
        acc11 = wmma_bf16(a1, b1, acc11);
        __syncthreads();
    }

    // C/D layout: VGPR r -> (lanes 0-15: M=r, lanes 16-31: M=8+r), N = lane&15
    const int lm  = lane & 15;
    const int hi8 = (lane & 16) ? 8 : 0;

#pragma unroll
    for (int i = 0; i < 2; ++i) {
#pragma unroll
        for (int j = 0; j < 2; ++j) {
            v8f c = (i == 0) ? (j == 0 ? acc00 : acc01)
                             : (j == 0 ? acc10 : acc11);
#pragma unroll
            for (int r = 0; r < 8; ++r) {
                int m = m0 + wm + i * 16 + hi8 + r;
                int n = n0 + wn + j * 16 + lm;
                float y = c[r] + bias[n];
                if (mode == 0) {
                    // power-of-two dims -> shifts/masks, no idiv
                    int b = m >> 12, s = m & (MHA_S - 1);
                    int h = n >> 6,  dh = n & (MHA_DH - 1);
                    ((__bf16*)Yout)[(((size_t)b * MHA_H + h) * MHA_S + s) *
                                        MHA_DH + dh] = (__bf16)y;
                } else {
                    ((float*)Yout)[(size_t)m * N + n] = y;
                }
            }
        }
    }
}

// ---------------------------------------------------------------------------
// Flash attention: Q,K,V in [B*H, S, 64] bf16; output [B, S, H*64] bf16.
// 8 waves/block; each wave owns a 16-row Q block (128 rows per block).
// Per step: 32 keys staged via TDM (tensor_load_to_lds), 32 values staged
// transposed; QK^T and P*V on WMMA; online softmax with 16-lane shfl_xor
// reductions matching the C-tile row layout.
// ---------------------------------------------------------------------------
__global__ __launch_bounds__(256) void mha_flash_attn_kernel(
    const __bf16* __restrict__ Q, const __bf16* __restrict__ Kx,
    const __bf16* __restrict__ Vx, __bf16* __restrict__ O, float scale) {
    const int DH = MHA_DH, S = MHA_S, H = MHA_H;
    __shared__ __bf16 Kt[32 * 64];       // [t][d]  row stride 128B
    __shared__ __bf16 Vt[64 * 32];       // [dh][t] row stride 64B (transposed)
    __shared__ __bf16 Ps[8][16 * 32];    // per-wave P tile, row stride 64B

    const int tid  = threadIdx.x;
    const int lane = tid & 31;
    const int wave = tid >> 5;
    const int qBlocks = S / 128;
    const int bh = blockIdx.x / qBlocks;
    const int q0 = (blockIdx.x % qBlocks) * 128 + wave * 16;

    const __bf16* Qp = Q  + (size_t)bh * S * DH;
    const __bf16* Kp = Kx + (size_t)bh * S * DH;
    const __bf16* Vp = Vx + (size_t)bh * S * DH;

    // Q fragments held in registers for the whole K/V sweep
    v16bf qa0 = ld_fragA(Qp, 128, q0, 0, lane);   // d = 0..31
    v16bf qa1 = ld_fragA(Qp, 128, q0, 64, lane);  // d = 32..63

    float mi[8], li[8];
#pragma unroll
    for (int r = 0; r < 8; ++r) { mi[r] = -3.0e38f; li[r] = 0.0f; }
    v8f o0 = {}, o1 = {}, o2 = {}, o3 = {};

    const int hi8 = (lane & 16) ? 8 : 0;
    const int lm  = lane & 15;

#if MHA_TDM
    const unsigned int ldsKt = (unsigned int)(size_t)(&Kt[0]);
#else
    const int kr = tid >> 3;            // 0..31  (manual K staging)
    const int kc = (tid & 7) * 8;       // 0..56
#endif
    const int vt = tid & 31;            // V staging: source row
    const int vd = (tid >> 5) * 8;      // dh slice

    for (int t0 = 0; t0 < S; t0 += 32) {
#if MHA_TDM
        // Tensor Data Mover: DMA the 32x64 bf16 K tile into LDS.
        // D# group0: count=1 | lds_addr | global_addr | type=2
        // D# group1: data_size=2B, tensor_dim0=64, tensor_dim1=S,
        //            tile=64x32, tensor_dim0_stride=64 (ISA 8.3/8.4)
        if (wave == 0) {
            unsigned long long ga =
                (unsigned long long)(size_t)(Kp + (size_t)t0 * DH);
            u32x4 g0;
            g0[0] = 1u;                                   // count = 1
            g0[1] = ldsKt;                                // lds_addr
            g0[2] = (unsigned int)ga;                     // global_addr lo
            g0[3] = (unsigned int)(ga >> 32) | (2u << 30);// addr hi | type=2
            i32x8 g1;
            g1[0] = 0x00010000;          // data_size = 1 (2 bytes)
            g1[1] = MHA_DH << 16;        // tensor_dim0 = 64  (bits 79:48)
            g1[2] = (int)((unsigned)MHA_S << 16); // tensor_dim1 lo (bits 95:80)
            g1[3] = MHA_DH << 16;        // tile_dim0 = 64    (bits 127:112)
            g1[4] = 32;                  // tile_dim1 = 32
            g1[5] = MHA_DH;              // tensor_dim0_stride = 64
            g1[6] = 0;
            g1[7] = 0;
            i32x4 gz4 = {0, 0, 0, 0};
            i32x8 gz8 = {0, 0, 0, 0, 0, 0, 0, 0};
            // 6-arg form (clang-23 / therock-10.0 lane):
            //   (g0, g1, g2, g3, g4, cpol)
            __builtin_amdgcn_tensor_load_to_lds(g0, g1, gz4, gz4, gz8, 0);
#if __has_builtin(__builtin_amdgcn_s_wait_tensorcnt)
            __builtin_amdgcn_s_wait_tensorcnt(0);
#else
            asm volatile("s_wait_tensorcnt 0x0" ::: "memory");
#endif
        }
#else
        *(uint4*)(Kt + kr * 64 + kc) =
            *(const uint4*)(Kp + (size_t)(t0 + kr) * DH + kc);
#endif
        // stage V tile transposed: Vt[dh][t]
        {
            const __bf16* src = Vp + (size_t)(t0 + vt) * DH + vd;
#pragma unroll
            for (int j = 0; j < 8; ++j) Vt[(vd + j) * 32 + vt] = src[j];
        }
        __syncthreads();

        // S = Q * K^T : two 16x16 score tiles (t-chunks), contraction d=64
        v8f sc0 = {}, sc1 = {};
        sc0 = wmma_bf16(qa0, ld_fragB(Kt, 128, 0, 0, lane), sc0);
        sc0 = wmma_bf16(qa1, ld_fragB(Kt, 128, 0, 64, lane), sc0);
        sc1 = wmma_bf16(qa0, ld_fragB(Kt, 128, 16, 0, lane), sc1);
        sc1 = wmma_bf16(qa1, ld_fragB(Kt, 128, 16, 64, lane), sc1);

        // online softmax, per output row (row = r + hi8 for this lane)
#pragma unroll
        for (int r = 0; r < 8; ++r) {
            float x0 = sc0[r] * scale;
            float x1 = sc1[r] * scale;
            float rmax = fmaxf(x0, x1);
#pragma unroll
            for (int m = 8; m >= 1; m >>= 1)
                rmax = fmaxf(rmax, __shfl_xor(rmax, m, 32));
            float mnew  = fmaxf(mi[r], rmax);
            float alpha = __expf(mi[r] - mnew);
            mi[r] = mnew;
            float p0 = __expf(x0 - mnew);
            float p1 = __expf(x1 - mnew);
            float psum = p0 + p1;
#pragma unroll
            for (int m = 8; m >= 1; m >>= 1)
                psum += __shfl_xor(psum, m, 32);
            li[r] = li[r] * alpha + psum;
            o0[r] *= alpha; o1[r] *= alpha; o2[r] *= alpha; o3[r] *= alpha;
            // spill P (bf16) into wave-private LDS, row-major 16x32
            Ps[wave][(r + hi8) * 32 + lm]      = (__bf16)p0;
            Ps[wave][(r + hi8) * 32 + 16 + lm] = (__bf16)p1;
        }

        // O += P * V  (contraction over 32 keys, 4 dh-chunks of 16)
        v16bf pa = ld_fragA(Ps[wave], 64, 0, 0, lane);
        o0 = wmma_bf16(pa, ld_fragB(Vt, 64, 0, 0, lane), o0);
        o1 = wmma_bf16(pa, ld_fragB(Vt, 64, 16, 0, lane), o1);
        o2 = wmma_bf16(pa, ld_fragB(Vt, 64, 32, 0, lane), o2);
        o3 = wmma_bf16(pa, ld_fragB(Vt, 64, 48, 0, lane), o3);
        __syncthreads();
    }

    // normalize + scatter into [B, S, H*64]
    const int b = bh >> 4;          // H = 16
    const int h = bh & (H - 1);
    __bf16* Ob = O + ((size_t)b * S) * (H * DH);
#pragma unroll
    for (int c = 0; c < 4; ++c) {
        v8f oc = (c == 0) ? o0 : (c == 1) ? o1 : (c == 2) ? o2 : o3;
#pragma unroll
        for (int r = 0; r < 8; ++r) {
            int m = q0 + r + hi8;
            int n = h * DH + c * 16 + lm;
            Ob[(size_t)m * (H * DH) + n] = (__bf16)(oc[r] / li[r]);
        }
    }
}

// ---------------------------------------------------------------------------
// host launcher
// ---------------------------------------------------------------------------
extern "C" void kernel_launch(void* const* d_in, const int* in_sizes, int n_in,
                              void* d_out, int out_size, void* d_ws,
                              size_t ws_size, hipStream_t stream) {
    (void)n_in; (void)out_size; (void)ws_size;
    const int D = MHA_D, H = MHA_H, S = MHA_S;
    const int BS = in_sizes[0] / D;   // B * S
    const int B  = BS / S;

    const float* q   = (const float*)d_in[0];
    const float* k   = (const float*)d_in[1];
    const float* v   = (const float*)d_in[2];
    const float* w_q = (const float*)d_in[3];
    const float* b_q = (const float*)d_in[4];
    const float* w_k = (const float*)d_in[5];
    const float* b_k = (const float*)d_in[6];
    const float* w_v = (const float*)d_in[7];
    const float* b_v = (const float*)d_in[8];
    const float* w_o = (const float*)d_in[9];
    const float* b_o = (const float*)d_in[10];

    size_t off = 0;
    auto carve = [&](size_t bytes) -> void* {
        void* p = (char*)d_ws + off;
        off += (bytes + 255) & ~(size_t)255;
        return p;
    };
    const size_t actB = (size_t)BS * D * sizeof(__bf16);
    const size_t wB   = (size_t)D * D * sizeof(__bf16);
    __bf16* qb  = (__bf16*)carve(actB);
    __bf16* kb  = (__bf16*)carve(actB);
    __bf16* vb  = (__bf16*)carve(actB);
    __bf16* wqb = (__bf16*)carve(wB);
    __bf16* wkb = (__bf16*)carve(wB);
    __bf16* wvb = (__bf16*)carve(wB);
    __bf16* wob = (__bf16*)carve(wB);
    __bf16* Qh  = (__bf16*)carve(actB);   // [B,H,S,Dh]
    __bf16* Kh  = (__bf16*)carve(actB);
    __bf16* Vh  = (__bf16*)carve(actB);
    __bf16* Ab  = (__bf16*)carve(actB);   // attention output [B,S,D]

    // 1) convert everything to bf16 (vectorized x4)
    const int CT = 256, CG = 1024;
    mha_cvt_bf16_kernel<<<CG, CT, 0, stream>>>((const float4*)q, qb,
                                               (size_t)BS * D / 4);
    mha_cvt_bf16_kernel<<<CG, CT, 0, stream>>>((const float4*)k, kb,
                                               (size_t)BS * D / 4);
    mha_cvt_bf16_kernel<<<CG, CT, 0, stream>>>((const float4*)v, vb,
                                               (size_t)BS * D / 4);
    mha_cvt_bf16_kernel<<<CG, CT, 0, stream>>>((const float4*)w_q, wqb,
                                               (size_t)D * D / 4);
    mha_cvt_bf16_kernel<<<CG, CT, 0, stream>>>((const float4*)w_k, wkb,
                                               (size_t)D * D / 4);
    mha_cvt_bf16_kernel<<<CG, CT, 0, stream>>>((const float4*)w_v, wvb,
                                               (size_t)D * D / 4);
    mha_cvt_bf16_kernel<<<CG, CT, 0, stream>>>((const float4*)w_o, wob,
                                               (size_t)D * D / 4);

    // 2) Q/K/V projections into head layout
    dim3 ggrid((BS / GM_BM) * (D / GM_BN));
    mha_gemm_xwt_kernel<<<ggrid, 256, 0, stream>>>(qb, wqb, b_q, Qh, BS, D, D, 0);
    mha_gemm_xwt_kernel<<<ggrid, 256, 0, stream>>>(kb, wkb, b_k, Kh, BS, D, D, 0);
    mha_gemm_xwt_kernel<<<ggrid, 256, 0, stream>>>(vb, wvb, b_v, Vh, BS, D, D, 0);

    // 3) flash attention
    dim3 agrid(B * H * (S / 128));
    mha_flash_attn_kernel<<<agrid, 256, 0, stream>>>(Qh, Kh, Vh, Ab,
                                                     0.125f /* 1/sqrt(64) */);

    // 4) output projection -> f32 d_out
    mha_gemm_xwt_kernel<<<ggrid, 256, 0, stream>>>(Ab, wob, b_o, d_out, BS, D,
                                                   D, 1);
}
